// Downsample_Block_72739566125667
// MI455X (gfx1250) — matrix operand order repaired and checked by reading, not compiled, not run on hardware
//
#include <hip/hip_runtime.h>

// ---------------------------------------------------------------------------
// Downsample block (1x1 stride-2 conv + folded-BN int8 QAT) as a WMMA GEMM.
//   M = Cout = 1024, K = Cin = 512, N = batch*14*14 = 12544
// Pipeline:
//   k1: fold BN into weights, per-channel int8 fake-quant, fold sign(gamma),
//       emit f16 A[M][K] and f32 bias[M]
//   k2: gather stride-2 activations into f16 Bp[N][K] (K contiguous)
//   k3: tiled wave32 WMMA f32_16x16x32_f16 GEMM + fused output fake-quant
// Staging in k3 uses gfx1250 async global->LDS loads (ASYNCcnt path).
// ---------------------------------------------------------------------------

typedef _Float16 v8h  __attribute__((ext_vector_type(8)));
typedef _Float16 v16h __attribute__((ext_vector_type(16)));
typedef float    v8f  __attribute__((ext_vector_type(8)));
typedef int      v4i  __attribute__((__vector_size__(4 * sizeof(int))));

#define CIN   512
#define COUT  1024
#define HIN   28
#define HOUT  14
#define NBATCH 64
#define NTOT  (NBATCH * HOUT * HOUT)   // 12544 = 98 * 128

#define BM 128
#define BN 128
#define BK 32
#define LDA 40            // BK + 8 halves pad -> 80B row stride, bank-conflict free

#if defined(__gfx1250__) && \
    __has_builtin(__builtin_amdgcn_global_load_async_to_lds_b128) && \
    __has_builtin(__builtin_amdgcn_s_wait_asynccnt)
#define USE_ASYNC_LDS 1
#else
#define USE_ASYNC_LDS 0
#endif

// builtin signature: (v4i addrspace(1)* src, v4i addrspace(3)* dst, imm offset, imm cpol)
#define GLOBAL_V4I(p) ((__attribute__((address_space(1))) v4i*)(p))
#define LDS_V4I(p)    ((__attribute__((address_space(3))) v4i*)(p))

__device__ __forceinline__ v16h cat8(v8h lo, v8h hi) {
    return __builtin_shufflevector(lo, hi, 0,1,2,3,4,5,6,7,8,9,10,11,12,13,14,15);
}

// --------------------------- kernel 1: weights ------------------------------
// One block per output channel. Fold BN scale, int8 symmetric quant, fold sign.
__global__ __launch_bounds__(256) void wq_kernel(
    const float* __restrict__ w, const float* __restrict__ gamma,
    const float* __restrict__ beta, const float* __restrict__ rmean,
    const float* __restrict__ rvar, _Float16* __restrict__ A,
    float* __restrict__ bias)
{
    const int co = blockIdx.x;
    const int t  = threadIdx.x;        // 256 threads, 512 weights per channel

    const float g       = gamma[co];
    const float inv_std = rsqrtf(rvar[co] + 1e-5f);
    const float fact    = fabsf(g) * inv_std;

    const float w0 = w[(size_t)co * CIN + t]       * fact;
    const float w1 = w[(size_t)co * CIN + t + 256] * fact;

    __shared__ float red[256];
    red[t] = fmaxf(fabsf(w0), fabsf(w1));
    __syncthreads();
    #pragma unroll
    for (int s = 128; s > 0; s >>= 1) {
        if (t < s) red[t] = fmaxf(red[t], red[t + s]);
        __syncthreads();
    }
    const float delta = fmaxf(red[0], 1e-8f) * (1.0f / 127.0f);
    const float sgn   = (g > 0.f) ? 1.f : ((g < 0.f) ? -1.f : 0.f);

    const float q0 = fminf(fmaxf(rintf(w0 / delta), -127.f), 127.f) * delta * sgn;
    const float q1 = fminf(fmaxf(rintf(w1 / delta), -127.f), 127.f) * delta * sgn;

    A[(size_t)co * CIN + t]       = (_Float16)q0;
    A[(size_t)co * CIN + t + 256] = (_Float16)q1;

    if (t == 0) bias[co] = beta[co] - rmean[co] * g * inv_std;
}

// ------------------------ kernel 2: activation pack -------------------------
// Gather x[b, k, 2h, 2w] -> Bp[n][k] (f16, K contiguous), n = b*196 + h*14 + w
__global__ __launch_bounds__(256) void pack_kernel(
    const float* __restrict__ x, _Float16* __restrict__ Bp)
{
    const int n   = blockIdx.x;             // 0..12543
    const int b   = n / (HOUT * HOUT);
    const int rem = n % (HOUT * HOUT);
    const int h   = rem / HOUT;
    const int wcol= rem % HOUT;

    const float* __restrict__ xp =
        x + (size_t)b * CIN * HIN * HIN + (size_t)(2 * h) * HIN + (2 * wcol);
    _Float16* __restrict__ dst = Bp + (size_t)n * CIN;

    #pragma unroll
    for (int i = 0; i < 2; ++i) {
        const int k = threadIdx.x + i * 256;
        dst[k] = (_Float16)xp[(size_t)k * (HIN * HIN)];
    }
}

// --------------------------- kernel 3: WMMA GEMM ----------------------------
__global__ __launch_bounds__(256) void gemm_kernel(
    const _Float16* __restrict__ A,       // [COUT][CIN]
    const _Float16* __restrict__ Bp,      // [NTOT][CIN]
    const float* __restrict__ bias,       // [COUT]
    const float* __restrict__ odelta,     // [COUT]
    float* __restrict__ out)              // [NBATCH][COUT][HOUT][HOUT]
{
    __shared__ _Float16 sA[BM * LDA];     // [m][k] padded
    __shared__ _Float16 sB[BN * LDA];     // [n][k] padded

    const int t     = threadIdx.x;
    const int lane  = t & 31;
    const int wave  = t >> 5;             // 8 waves: 2 (M) x 4 (N)
    const int waveM = wave >> 2;
    const int waveN = wave & 3;
    const int l16   = lane & 15;
    const int hsel  = lane >> 4;          // 0: lanes 0-15, 1: lanes 16-31

    const int bm = blockIdx.x * BM;       // gridDim.x = 8
    const int bn = blockIdx.y * BN;       // gridDim.y = 98

    v8f acc[4][2] = {};

    for (int k0 = 0; k0 < CIN; k0 += BK) {
        // --- stage tiles: 256 threads x 2 chunks of 16B each per matrix ---
        #pragma unroll
        for (int i = 0; i < 2; ++i) {
            const int c   = t * 2 + i;          // 0..511
            const int row = c >> 2;             // 0..127
            const int kk  = (c & 3) * 8;        // 0,8,16,24 halves
            const _Float16* gA = &A [(size_t)(bm + row) * CIN + k0 + kk];
            const _Float16* gB = &Bp[(size_t)(bn + row) * CIN + k0 + kk];
            _Float16* lA = &sA[row * LDA + kk];
            _Float16* lB = &sB[row * LDA + kk];
#if USE_ASYNC_LDS
            // gfx1250 direct global->LDS async copy (ASYNCcnt), 16B per lane
            __builtin_amdgcn_global_load_async_to_lds_b128(
                GLOBAL_V4I(gA), LDS_V4I(lA), 0, 0);
            __builtin_amdgcn_global_load_async_to_lds_b128(
                GLOBAL_V4I(gB), LDS_V4I(lB), 0, 0);
#else
            *(v8h*)lA = *(const v8h*)gA;
            *(v8h*)lB = *(const v8h*)gB;
#endif
            if (k0 + BK < CIN) {                // hint next K-slab into L2
                __builtin_prefetch(&A [(size_t)(bm + row) * CIN + k0 + BK + kk], 0, 1);
                __builtin_prefetch(&Bp[(size_t)(bn + row) * CIN + k0 + BK + kk], 0, 1);
            }
        }
#if USE_ASYNC_LDS
        __builtin_amdgcn_s_wait_asynccnt(0);
#endif
        __syncthreads();

        // --- compute: 8 x v_wmma_f32_16x16x32_f16 per wave per K-slab ---
        #pragma unroll
        for (int nf = 0; nf < 2; ++nf) {
            // B frag (32x16 f16): lanes 0-15 col=l16 K0..7|16..23, lanes 16-31 K8..15|24..31
            const int brow = waveN * 32 + nf * 16 + l16;
            const v8h b0 = *(const v8h*)&sB[brow * LDA + hsel * 8];
            const v8h b1 = *(const v8h*)&sB[brow * LDA + 16 + hsel * 8];
            const v16h bfrag = cat8(b0, b1);

            #pragma unroll
            for (int mf = 0; mf < 4; ++mf) {
                // A frag (16x32 f16): lanes 0-15 K0..7|16..23, lanes 16-31 K8..15|24..31
                const int arow = waveM * 64 + mf * 16 + l16;
                const v8h a0 = *(const v8h*)&sA[arow * LDA + hsel * 8];
                const v8h a1 = *(const v8h*)&sA[arow * LDA + 16 + hsel * 8];
                const v16h afrag = cat8(a0, a1);

                acc[mf][nf] = __builtin_amdgcn_wmma_f32_16x16x32_f16(
                    /*neg_a=*/false, afrag, /*neg_b=*/false, bfrag,
                    /*c_mod=*/(short)0, acc[mf][nf],
                    /*reuse_a=*/false, /*reuse_b=*/false);
            }
        }
        __syncthreads();
    }

    // --- epilogue: bias (sign already folded into A) + per-channel out quant ---
    // C layout: VGPR j -> lanes 0-15: M=j, N=lane ; lanes 16-31: M=j+8, N=lane-16
    #pragma unroll
    for (int nf = 0; nf < 2; ++nf) {
        const int n   = bn + waveN * 32 + nf * 16 + l16;
        const int b   = n / (HOUT * HOUT);
        const int rem = n % (HOUT * HOUT);
        const int hh  = rem / HOUT;
        const int ww  = rem % HOUT;
        #pragma unroll
        for (int mf = 0; mf < 4; ++mf) {
            const int mbase = bm + waveM * 64 + mf * 16 + (hsel << 3);
            #pragma unroll
            for (int j = 0; j < 8; ++j) {
                const int m   = mbase + j;
                const float v = acc[mf][nf][j] + bias[m];
                const float od = odelta[m];
                const float q = fminf(fmaxf(rintf(v / od), -128.f), 127.f) * od;
                out[(((size_t)b * COUT + m) * HOUT + hh) * HOUT + ww] = q;
            }
        }
    }
}

// ------------------------------- launcher -----------------------------------
extern "C" void kernel_launch(void* const* d_in, const int* in_sizes, int n_in,
                              void* d_out, int out_size, void* d_ws, size_t ws_size,
                              hipStream_t stream)
{
    const float* x      = (const float*)d_in[0];   // (64,512,28,28)
    const float* weight = (const float*)d_in[1];   // (1024,512,1,1)
    const float* gamma  = (const float*)d_in[2];
    const float* beta   = (const float*)d_in[3];
    const float* rmean  = (const float*)d_in[4];
    const float* rvar   = (const float*)d_in[5];
    const float* odelta = (const float*)d_in[6];
    float* out = (float*)d_out;                    // (64,1024,14,14)

    // workspace carve-up
    char* ws = (char*)d_ws;
    _Float16* A    = (_Float16*)ws;                                  // 1 MB
    _Float16* Bp   = (_Float16*)(ws + (size_t)COUT * CIN * 2);       // ~12.25 MB
    float*    bias = (float*)(ws + (size_t)COUT * CIN * 2
                                 + (size_t)NTOT * CIN * 2);          // 4 KB

    wq_kernel<<<COUT, 256, 0, stream>>>(weight, gamma, beta, rmean, rvar, A, bias);
    pack_kernel<<<NTOT, 256, 0, stream>>>(x, Bp);

    dim3 grid(COUT / BM, NTOT / BN);   // 8 x 98
    gemm_kernel<<<grid, 256, 0, stream>>>(A, Bp, bias, odelta, out);
}
